// RDM_Model_81801947119980
// MI455X (gfx1250) — compile-verified, matrix operand order checked by reading, not run on hardware
//
#include <hip/hip_runtime.h>
#include <hip/hip_bf16.h>
#include <math.h>

typedef __attribute__((ext_vector_type(16))) __bf16 v16bf;
typedef __attribute__((ext_vector_type(8)))  float  v8f;
typedef int gv4i __attribute__((vector_size(16)));   // matches async builtin param type
typedef __attribute__((address_space(1))) gv4i* as1_v4i_p;
typedef __attribute__((address_space(3))) gv4i* as3_v4i_p;

#define B_    16
#define SENT  50
#define L_    300
#define SEQT  50
#define EMB   6
#define HID   256
#define G3    768
#define OUTD  2
#define NTILE 48   // 768 / 16 N-tiles
#define KCH   8    // 256 / 32 K-chunks
#define NTHR  512  // 16 waves = 4 waves/SIMD -> 256-VGPR budget per wave
#define TPW   3    // tiles per wave: 16 * 3 = 48

#if defined(__has_builtin)
#  if __has_builtin(__builtin_amdgcn_global_load_async_to_lds_b128)
#    define HAVE_ASYNC_LDS 1
#  endif
#  if __has_builtin(__builtin_amdgcn_s_wait_asynccnt)
#    define HAVE_WAIT_ASYNC 1
#  endif
#endif

__device__ __forceinline__ void wait_asynccnt0() {
#if defined(HAVE_WAIT_ASYNC)
  __builtin_amdgcn_s_wait_asynccnt(0);
#else
  asm volatile("s_wait_asynccnt 0" ::: "memory");
#endif
}

__device__ __forceinline__ unsigned short f2bf(float f) {
  unsigned int u = __float_as_uint(f);
  u += 0x7FFFu + ((u >> 16) & 1u);   // round-to-nearest-even
  return (unsigned short)(u >> 16);
}

union BFrag {
  v16bf bf;
  unsigned int u32[8];
  uint4 q[2];
};

// ---------------- pooling: seq[b, l] = max_o (Wp[o,:] . x[b,:,l] + bp[o]) ----------------
__global__ void pool_kernel(const float* __restrict__ x, const float* __restrict__ Wp,
                            const float* __restrict__ bp, float* __restrict__ seq) {
  int i = blockIdx.x * blockDim.x + threadIdx.x;
  if (i >= B_ * L_) return;
  int b = i / L_, l = i % L_;
  const float* xb = x + b * SENT * L_ + l;
  float m = -3.4e38f;
  for (int o = 0; o < SENT; ++o) {
    float acc = bp[o];
    const float* wo = Wp + o * SENT;
    for (int s = 0; s < SENT; ++s) acc += xb[s * L_] * wo[s];
    m = fmaxf(m, acc);
  }
  seq[i] = m;   // seq[b*300 + t*6 + e], l = t*6 + e
}

// ---------------- input projection: gi[bt, g] = bih[g] + seq[bt,:] . Wih[g,:] ----------------
__global__ void gi_kernel(const float* __restrict__ seq, const float* __restrict__ Wih,
                          const float* __restrict__ bih, float* __restrict__ gi) {
  int i = blockIdx.x * blockDim.x + threadIdx.x;
  if (i >= B_ * SEQT * G3) return;
  int g  = i % G3;
  int bt = i / G3;
  const float* s = seq + bt * EMB;
  const float* w = Wih + g * EMB;
  float acc = bih[g];
#pragma unroll
  for (int d = 0; d < EMB; ++d) acc += s[d] * w[d];
  gi[i] = acc;
}

// ---------------- pack Whh into bf16 WMMA B-fragment layout (ISA 7.12.2) ----------------
// frag index layout: ((tile*8 + c)*32 + lane)*8 + v  (one u32 = packed bf16 K-pair)
__global__ void whh_pack_kernel(const float* __restrict__ Whh, unsigned int* __restrict__ frag) {
  int i = blockIdx.x * blockDim.x + threadIdx.x;
  if (i >= NTILE * KCH * 32 * 8) return;
  int v    = i & 7;
  int lane = (i >> 3) & 31;
  int c    = (i >> 8) & 7;
  int tile = i >> 11;
  int half = lane >> 4;
  int p = (v & 3) + 4 * (((v >> 2) << 1) + half);   // K-pair index within 32-K chunk
  int k = c * 32 + 2 * p;
  int j = tile * 16 + (lane & 15);                  // gate (Whh row) index
  const float* src = Whh + j * HID + k;
  unsigned int lo = f2bf(src[0]);
  unsigned int hi = f2bf(src[1]);
  frag[i] = lo | (hi << 16);
}

// ---------------- persistent GRU + FC: one workgroup, 16 waves, 50 steps ----------------
__global__ void __launch_bounds__(NTHR)
gru_kernel(const float* __restrict__ gi, const unsigned int* __restrict__ whh_frag,
           const float* __restrict__ bhh, const float* __restrict__ Wfc,
           const float* __restrict__ bfc, float* __restrict__ out) {
  __shared__ __align__(16) unsigned short h_bf[B_ * HID];   // 8 KB: h as bf16 for WMMA A
  __shared__ __align__(16) float          h_f [B_ * HID];   // 16 KB: h in f32
  __shared__ __align__(16) float          gh_s[B_ * G3];    // 48 KB: gh = h . Whh^T
  __shared__ __align__(16) float          gi_s[B_ * G3];    // 48 KB: gi[:, t, :] staged
  __shared__ float                        bhh_s[G3];        // 3 KB
  __shared__ float                        wfc_s[OUTD * HID];// 2 KB
  __shared__ float                        bfc_s[OUTD];

  const int tid  = threadIdx.x;
  const int lane = tid & 31;
  const int wave = tid >> 5;
  const int half = lane >> 4;
  const int col  = lane & 15;

  for (int k = tid; k < B_ * HID; k += NTHR) { h_bf[k] = 0; h_f[k] = 0.0f; }
  for (int k = tid; k < G3; k += NTHR)         bhh_s[k] = bhh[k];
  for (int k = tid; k < OUTD * HID; k += NTHR) wfc_s[k] = Wfc[k];
  if (tid < OUTD) bfc_s[tid] = bfc[tid];

  // gi staging: each thread copies 24 consecutive floats (6 x b128) of gi[:, t, :]
  const int cp_b = tid >> 5;            // 32 threads per batch row
  const int cp_g = (tid & 31) * 24;

  // -- preload this wave's 3 N-tiles of Whh fragments into registers (stays resident) --
  BFrag Bres[TPW][KCH];
#pragma unroll
  for (int tt = 0; tt < TPW; ++tt) {
    int tile = wave * TPW + tt;
#pragma unroll
    for (int c = 0; c < KCH; ++c) {
      const uint4* src = (const uint4*)(whh_frag + (((tile * KCH + c) * 32 + lane) << 3));
      Bres[tt][c].q[0] = src[0];
      Bres[tt][c].q[1] = src[1];
    }
  }
  __syncthreads();

  for (int t = 0; t < SEQT; ++t) {
    // -- stage gi[:, t, :] into LDS; async path keeps the copy out of VGPRs --
    {
      char* src = (char*)(gi + (cp_b * SEQT + t) * G3 + cp_g);
      char* dst = (char*)(gi_s + cp_b * G3 + cp_g);
#if defined(HAVE_ASYNC_LDS)
#pragma unroll
      for (int q = 0; q < 6; ++q)
        __builtin_amdgcn_global_load_async_to_lds_b128(
            (as1_v4i_p)(src + 16 * q),
            (as3_v4i_p)(dst + 16 * q),
            0, 0);
#else
#pragma unroll
      for (int q = 0; q < 6; ++q)
        ((uint4*)dst)[q] = ((const uint4*)src)[q];
#endif
    }

    // -- each wave: 3 N-tiles of gh = h . Whh^T via v_wmma_f32_16x16x32_bf16 --
    v8f acc0 = {}, acc1 = {}, acc2 = {};
#pragma unroll
    for (int c = 0; c < KCH; ++c) {
      // A fragment (16x32 bf16 chunk of h) from LDS, ISA 7.12.2 layout
      BFrag A;
#pragma unroll
      for (int v = 0; v < 8; ++v) {
        int p = (v & 3) + 4 * (((v >> 2) << 1) + half);
        int k = c * 32 + 2 * p;
        A.u32[v] = *(const unsigned int*)&h_bf[col * HID + k];
      }
      acc0 = __builtin_amdgcn_wmma_f32_16x16x32_bf16(false, A.bf, false, Bres[0][c].bf,
                                                     (short)0, acc0, false, false);
      acc1 = __builtin_amdgcn_wmma_f32_16x16x32_bf16(false, A.bf, false, Bres[1][c].bf,
                                                     (short)0, acc1, false, false);
      acc2 = __builtin_amdgcn_wmma_f32_16x16x32_bf16(false, A.bf, false, Bres[2][c].bf,
                                                     (short)0, acc2, false, false);
    }
    {
      int n0 = wave * (16 * TPW) + col;   // tiles 3w, 3w+1, 3w+2 -> cols 16 apart
#pragma unroll
      for (int r = 0; r < 8; ++r) {
        int m = r + 8 * half;             // C/D layout: VGPR r -> M = r (+8 upper half)
        gh_s[m * G3 + n0]      = acc0[r];
        gh_s[m * G3 + n0 + 16] = acc1[r];
        gh_s[m * G3 + n0 + 32] = acc2[r];
      }
    }
#if defined(HAVE_ASYNC_LDS)
    wait_asynccnt0();               // this wave's gi copies have landed in LDS
#endif
    __syncthreads();

    // -- elementwise GRU gates over [16,256], gates read from LDS --
#pragma unroll 1
    for (int u = tid; u < B_ * HID; u += NTHR) {
      int b = u >> 8;
      int j = u & 255;
      float ir  = gi_s[b * G3 + j]       + gh_s[b * G3 + j]       + bhh_s[j];
      float iz  = gi_s[b * G3 + 256 + j] + gh_s[b * G3 + 256 + j] + bhh_s[256 + j];
      float hn  = gh_s[b * G3 + 512 + j] + bhh_s[512 + j];
      float in_ = gi_s[b * G3 + 512 + j];
      float r = 1.0f / (1.0f + __expf(-ir));
      float z = 1.0f / (1.0f + __expf(-iz));
      float n = tanhf(in_ + r * hn);
      float hv = (1.0f - z) * n + z * h_f[u];
      h_f[u]  = hv;
      h_bf[u] = f2bf(hv);
    }
    __syncthreads();

    // -- FC 256->2 on fresh hidden state: one wave per batch row, shuffle reduction --
    {
      const int b = wave;               // 16 waves <-> 16 batch rows
#pragma unroll
      for (int o = 0; o < OUTD; ++o) {
        float acc = 0.0f;
#pragma unroll 4
        for (int j = lane; j < HID; j += 32) acc += wfc_s[o * HID + j] * h_f[b * HID + j];
#pragma unroll
        for (int off = 16; off > 0; off >>= 1) acc += __shfl_xor(acc, off, 32);
        if (lane == 0) out[(b * SEQT + t) * OUTD + o] = bfc_s[o] + acc;
      }
    }
    __syncthreads();
  }
}

extern "C" void kernel_launch(void* const* d_in, const int* in_sizes, int n_in,
                              void* d_out, int out_size, void* d_ws, size_t ws_size,
                              hipStream_t stream) {
  const float* x   = (const float*)d_in[0];
  const float* Wp  = (const float*)d_in[1];
  const float* bp  = (const float*)d_in[2];
  const float* Wih = (const float*)d_in[3];
  const float* Whh = (const float*)d_in[4];
  const float* bih = (const float*)d_in[5];
  const float* bhh = (const float*)d_in[6];
  const float* Wfc = (const float*)d_in[7];
  const float* bfc = (const float*)d_in[8];
  float* out = (float*)d_out;

  char* ws = (char*)d_ws;
  float*        seq  = (float*)ws;                            // 4800 f32
  float*        gi   = (float*)(ws + 19456);                  // 614400 f32
  unsigned int* frag = (unsigned int*)(ws + 19456 + 2457600); // 98304 u32 (bf16 Whh fragments)

  pool_kernel<<<(B_ * L_ + 255) / 256, 256, 0, stream>>>(x, Wp, bp, seq);
  whh_pack_kernel<<<(NTILE * KCH * 32 * 8 + 255) / 256, 256, 0, stream>>>(Whh, frag);
  gi_kernel<<<(B_ * SEQT * G3 + 255) / 256, 256, 0, stream>>>(seq, Wih, bih, gi);
  gru_kernel<<<1, NTHR, 0, stream>>>(gi, frag, bhh, Wfc, bfc, out);
}
